// SparseMessagePassing_22265110463271
// MI455X (gfx1250) — matrix operand order repaired and checked by reading, not compiled, not run on hardware
//
#include <hip/hip_runtime.h>

// ---------------------------------------------------------------------------
// SparseMessagePassing for MI455X (gfx1250): f16 WMMA GEMMs + f16 L2 gather
// ---------------------------------------------------------------------------

typedef __attribute__((ext_vector_type(16))) _Float16 v16h;
typedef __attribute__((ext_vector_type(8)))  _Float16 v8h;
typedef __attribute__((ext_vector_type(4)))  _Float16 v4h;
typedef __attribute__((ext_vector_type(8)))  float    v8f;

#define MTOT 8192   // B*S
#define SEQ  4096   // S
#define DDIM 512    // D
#define KNBR 32     // K neighbors

// ---- fragment loads (contiguous k order, same in A and B) ----------

// A-matrix 16x32 f16: lane[0..15] row M=lane&15 holds K = 0..15 contiguous;
// lanes 16..31 hold K = 16..31. p points at row*lda + khalf.
__device__ __forceinline__ v16h load_a_frag(const _Float16* p) {
  union { v16h v; v8h h[2]; } u;
  u.h[0] = *(const v8h*)(p);        // K = khalf .. khalf+7
  u.h[1] = *(const v8h*)(p + 8);    // K = khalf+8 .. khalf+15
  return u.v;
}

// B-matrix 32x16 f16: lane[0..15] col N=lane&15 holds K=0..15 contiguous;
// lanes 16..31 hold K=16..31. p points at col*ldb + kbase.
__device__ __forceinline__ v16h load_b_frag(const _Float16* p) {
  union { v16h v; v8h h[2]; } u;
  u.h[0] = *(const v8h*)(p);        // K = kbase .. kbase+7
  u.h[1] = *(const v8h*)(p + 8);    // K = kbase+8 .. kbase+15
  return u.v;
}

__device__ __forceinline__ v8f wmma_f16(v16h a, v16h b, v8f c) {
  // 8 args: (neg_a, A, neg_b, B, c_mod, C, reuse_a, reuse_b)
  return __builtin_amdgcn_wmma_f32_16x16x32_f16(false, a, false, b,
                                                (short)0, c, false, false);
}

// ---- epilogues --------------------------------------------------------------
// C/D layout: lane[0..15] -> N = lane, M = r;  lane[16..31] -> N = lane-16, M = 8+r
// EPI 0: out_f32 = acc + bias                         (final projection)
// EPI 1: out_f32 = acc + bias AND f16 copy            (agg: f32 for blend + concat half)
// EPI 2: g = sigmoid(acc+bias); f16 = x*(1-g)+agg*g   (fused gate+blend)
// EPI 3: out_f16 = acc + bias                         (messages, feeds f16 gather)
template <int EPI>
__device__ __forceinline__ void store_frag(
    v8f acc, int m0, int n0, int lane,
    const float* __restrict__ bias,
    float* __restrict__ outF, int ldcF,
    _Float16* __restrict__ outH, int ldcH,
    const float* __restrict__ xin, const float* __restrict__ aggf)
{
  const int cn = n0 + (lane & 15);
  const int cm = m0 + ((lane >> 4) << 3);
  const float bv = bias[cn];
#pragma unroll
  for (int r = 0; r < 8; ++r) {
    const float v = acc[r] + bv;
    const size_t row = (size_t)(cm + r);
    if (EPI == 0) {
      outF[row * (size_t)ldcF + cn] = v;
    } else if (EPI == 1) {
      outF[row * (size_t)ldcF + cn] = v;
      outH[row * (size_t)ldcH + cn] = (_Float16)v;
    } else if (EPI == 2) {
      const float g  = 1.0f / (1.0f + __expf(-v));
      const float xv = xin[row * (size_t)DDIM + cn];
      const float av = aggf[row * (size_t)DDIM + cn];
      outH[row * (size_t)ldcH + cn] = (_Float16)(xv * (1.0f - g) + av * g);
    } else {
      outH[row * (size_t)ldcH + cn] = (_Float16)v;
    }
  }
}

// ---- WMMA GEMM: out = A(MxKdim) * W^T(+bias), W is (N x Kdim) row-major ----
// Block: 256 threads = 8 waves as 4(M) x 2(N); wave tile 32x64 (2x4 frags,
// 8 WMMAs per 6 fragment loads). Block tile 128(M) x 128(N).
// Grid: (MTOT/128, N/128). All dims divide evenly -> no tails.
template <int EPI>
__global__ __launch_bounds__(256) void wmma_gemm(
    const _Float16* __restrict__ A, int lda,
    const _Float16* __restrict__ Wh, int ldb,   // ldb == Kdim
    const float* __restrict__ bias, int Kdim,
    float* __restrict__ outF, int ldcF,
    _Float16* __restrict__ outH, int ldcH,
    const float* __restrict__ xin, const float* __restrict__ aggf)
{
  const int lane = threadIdx.x & 31;
  const int wave = threadIdx.x >> 5;
  const int mb = blockIdx.x * 128 + (wave & 3) * 32;   // 4 waves in M
  const int nb = blockIdx.y * 128 + (wave >> 2) * 64;  // 2 waves in N

  const int r     = lane & 15;
  const int ahalf = (lane >> 4) << 4;   // 0 or 16
  const int bbase = (lane >> 4) << 4;   // 0 or 16

  const _Float16* ap[2];
  const _Float16* bp[4];
#pragma unroll
  for (int i = 0; i < 2; ++i)
    ap[i] = A + (size_t)(mb + 16 * i + r) * lda + ahalf;
#pragma unroll
  for (int j = 0; j < 4; ++j)
    bp[j] = Wh + (size_t)(nb + 16 * j + r) * ldb + bbase;

  v8f c[2][4];
#pragma unroll
  for (int i = 0; i < 2; ++i)
#pragma unroll
    for (int j = 0; j < 4; ++j)
      c[i][j] = (v8f){};

#pragma unroll 2
  for (int k = 0; k < Kdim; k += 32) {
    v16h af[2], bf[4];
#pragma unroll
    for (int i = 0; i < 2; ++i) af[i] = load_a_frag(ap[i] + k);
#pragma unroll
    for (int j = 0; j < 4; ++j) bf[j] = load_b_frag(bp[j] + k);
#pragma unroll
    for (int i = 0; i < 2; ++i)
#pragma unroll
      for (int j = 0; j < 4; ++j)
        c[i][j] = wmma_f16(af[i], bf[j], c[i][j]);
  }

#pragma unroll
  for (int i = 0; i < 2; ++i)
#pragma unroll
    for (int j = 0; j < 4; ++j)
      store_frag<EPI>(c[i][j], mb + 16 * i, nb + 16 * j, lane,
                      bias, outF, ldcF, outH, ldcH, xin, aggf);
}

// ---- gather + weighted neighbor sum (f16 rows, L2-resident) ----------------
// One block (128 threads) per token; each lane loads 4 halves (8B) per row ->
// one f16 row (1KB) per neighbor, f32 accumulate, f16 result feeds GEMM2
// directly. Topology/weight loads are block-uniform -> scalarized.
__global__ __launch_bounds__(128) void gather_agg_f16(
    const _Float16* __restrict__ msgh, const int* __restrict__ topo,
    const float* __restrict__ wts, _Float16* __restrict__ aggh)
{
  const int m = blockIdx.x;              // global token id
  const int b = m >> 12;                 // batch = m / SEQ
  const int* tp = topo + (size_t)m * KNBR;
  const float* wp = wts + (size_t)m * KNBR;
  const _Float16* base = msgh + (size_t)b * SEQ * DDIM;

  const int d4 = threadIdx.x;            // v4h index: 128 * 4 = 512
  float acc[4] = {0.f, 0.f, 0.f, 0.f};
#pragma unroll 8
  for (int k = 0; k < KNBR; ++k) {
    const int   idx = tp[k];
    const float w   = wp[k];
    const v4h v = ((const v4h*)(base + (size_t)idx * DDIM))[d4];
#pragma unroll
    for (int j = 0; j < 4; ++j)
      acc[j] = fmaf(w, (float)v[j], acc[j]);
  }
  v4h o;
#pragma unroll
  for (int j = 0; j < 4; ++j) o[j] = (_Float16)acc[j];
  ((v4h*)(aggh + (size_t)m * DDIM))[d4] = o;
}

// ---- conversions ------------------------------------------------------------
__global__ __launch_bounds__(256) void cvt_f32_to_f16(
    const float* __restrict__ in, _Float16* __restrict__ out, int n)
{
  const int i = blockIdx.x * 256 + threadIdx.x;
  if (i < n) out[i] = (_Float16)in[i];
}

// x (M x 512, dense) -> first half of concat buffer (stride 1024)
__global__ __launch_bounds__(256) void cvt_x_to_cat(
    const float* __restrict__ x, _Float16* __restrict__ cat)
{
  const int i = blockIdx.x * 256 + threadIdx.x;  // over MTOT*DDIM
  const int m = i >> 9;
  const int d = i & 511;
  cat[(size_t)m * 1024 + d] = (_Float16)x[i];
}

// ---------------------------------------------------------------------------
extern "C" void kernel_launch(void* const* d_in, const int* in_sizes, int n_in,
                              void* d_out, int out_size, void* d_ws, size_t ws_size,
                              hipStream_t stream) {
  const float* x    = (const float*)d_in[0];
  const int*   topo = (const int*)  d_in[1];
  const float* wts  = (const float*)d_in[2];
  const float* Wm   = (const float*)d_in[3];
  const float* bm   = (const float*)d_in[4];
  const float* Wa   = (const float*)d_in[5];
  const float* ba   = (const float*)d_in[6];
  const float* Wg   = (const float*)d_in[7];
  const float* bg   = (const float*)d_in[8];
  const float* Wo   = (const float*)d_in[9];
  const float* bo   = (const float*)d_in[10];

  // workspace carve (all regions fully written before being read)
  char* ws = (char*)d_ws;
  size_t off = 0;
  auto carve = [&](size_t bytes) {
    char* p = ws + off;
    off += (bytes + 255) & ~(size_t)255;
    return p;
  };
  _Float16* cat  = (_Float16*)carve((size_t)MTOT * 1024 * sizeof(_Float16)); // [xh | aggh]
  _Float16* msgh = (_Float16*)carve((size_t)MTOT * DDIM * sizeof(_Float16)); // f16 messages
  _Float16* aggh = (_Float16*)carve((size_t)MTOT * DDIM * sizeof(_Float16)); // gather out (GEMM2 A)
  float*    aggf = (float*)   carve((size_t)MTOT * DDIM * sizeof(float));    // f32 agg for blend
  _Float16* Wmh  = (_Float16*)carve((size_t)DDIM * DDIM * sizeof(_Float16));
  _Float16* Wah  = (_Float16*)carve((size_t)DDIM * DDIM * sizeof(_Float16));
  _Float16* Woh  = (_Float16*)carve((size_t)DDIM * DDIM * sizeof(_Float16));
  _Float16* Wgh  = (_Float16*)carve((size_t)DDIM * 1024 * sizeof(_Float16));
  _Float16* updh = msgh;   // reuse: messages dead after the gather

  const dim3 blk(256);
  const dim3 ggrid(MTOT / 128, DDIM / 128);

  // 1) precision conversion of weights + x
  cvt_f32_to_f16<<<(DDIM * DDIM + 255) / 256, blk, 0, stream>>>(Wm, Wmh, DDIM * DDIM);
  cvt_f32_to_f16<<<(DDIM * DDIM + 255) / 256, blk, 0, stream>>>(Wa, Wah, DDIM * DDIM);
  cvt_f32_to_f16<<<(DDIM * DDIM + 255) / 256, blk, 0, stream>>>(Wo, Woh, DDIM * DDIM);
  cvt_f32_to_f16<<<(DDIM * 1024 + 255) / 256, blk, 0, stream>>>(Wg, Wgh, DDIM * 1024);
  cvt_x_to_cat<<<MTOT * DDIM / 256, blk, 0, stream>>>(x, cat);

  // 2) messages = x @ Wm.T + bm  (f16 out -> halves gather traffic)
  wmma_gemm<3><<<ggrid, blk, 0, stream>>>(cat, 1024, Wmh, DDIM, bm, DDIM,
                                          nullptr, 0, msgh, DDIM, nullptr, nullptr);

  // 3) weighted neighbor gather (L2-bound, f16 rows, f32 accumulate)
  gather_agg_f16<<<MTOT, dim3(128), 0, stream>>>(msgh, topo, wts, aggh);

  // 4) agg = gathered @ Wa.T + ba; epilogue emits f32 agg (for gate blend)
  //    plus f16 copy into the concat buffer's second half
  wmma_gemm<1><<<ggrid, blk, 0, stream>>>(aggh, DDIM, Wah, DDIM, ba, DDIM,
                                          aggf, DDIM, cat + DDIM, 1024, nullptr, nullptr);

  // 5) gate GEMM over [x|agg] (Kdim=1024) with fused sigmoid-gate-blend epilogue
  wmma_gemm<2><<<ggrid, blk, 0, stream>>>(cat, 1024, Wgh, 1024, bg, 1024,
                                          nullptr, 0, updh, DDIM, x, aggf);

  // 6) out = updated @ Wo.T + bo  (f32 to d_out)
  wmma_gemm<0><<<ggrid, blk, 0, stream>>>(updh, DDIM, Woh, DDIM, bo, DDIM,
                                          (float*)d_out, DDIM, nullptr, 0, nullptr, nullptr);
}
